// Block_51178830299350
// MI455X (gfx1250) — matrix operand order repaired and checked by reading, not compiled, not run on hardware
//
#include <hip/hip_runtime.h>
#include <hip/hip_bf16.h>
#include <math.h>

// ---------------------------------------------------------------------------
// Types for CDNA5 WMMA (wave32): v_wmma_f32_16x16x32_bf16
// ---------------------------------------------------------------------------
typedef __bf16 bf16_t;
typedef __attribute__((ext_vector_type(16))) __bf16 v16bf;
typedef __attribute__((ext_vector_type(8)))  __bf16 v8bf;
typedef __attribute__((ext_vector_type(8)))  float   v8f;

#define BB 2
#define SS 2048
#define DD 1024
#define HH 16
#define HDIM 64
#define FFN 4096
#define MTOT (BB*SS)          // 4096 rows of activations

// round-to-nearest-even fp32 -> bf16 (bit math; no libcalls)
__device__ __forceinline__ bf16_t f2bf(float f) {
  unsigned u = __builtin_bit_cast(unsigned, f);
  u += 0x7FFFu + ((u >> 16) & 1u);
  unsigned short h = (unsigned short)(u >> 16);
  return __builtin_bit_cast(bf16_t, h);
}

__device__ __forceinline__ v8f zero8() {
  v8f z = {0.f,0.f,0.f,0.f,0.f,0.f,0.f,0.f};
  return z;
}

// A-fragment (16xK bf16): lane holds row m = lane%16; element e maps to
// K = t*32 + ((e<8) ? g*8+e : 16+g*8+(e-8)), g = lane/16  (ISA 7.12.2)
__device__ __forceinline__ v16bf load_frag_a(const bf16_t* row, int g, int t) {
  const v8bf lo = *(const v8bf*)(row + t*32 + g*8);
  const v8bf hi = *(const v8bf*)(row + t*32 + 16 + g*8);
  return __builtin_shufflevector(lo, hi, 0,1,2,3,4,5,6,7,8,9,10,11,12,13,14,15);
}

// B-fragment (Kx16 bf16): lane holds col n = lane%16; element e maps to
// K = t*32 + ((e<8) ? g*8+e : 16+g*8+(e-8)). `col` points to contiguous-K storage of column n.
__device__ __forceinline__ v16bf load_frag_b(const bf16_t* col, int g, int t) {
  const v8bf lo = *(const v8bf*)(col + t*32 + g*8);
  const v8bf hi = *(const v8bf*)(col + t*32 + 16 + g*8);
  return __builtin_shufflevector(lo, hi, 0,1,2,3,4,5,6,7,8,9,10,11,12,13,14,15);
}

__device__ __forceinline__ v8f wmma_bf16(v16bf a, v16bf b, v8f c) {
  return __builtin_amdgcn_wmma_f32_16x16x32_bf16(false, a, false, b, (short)0, c, false, false);
}

// ---------------------------------------------------------------------------
// CDNA5 async global->LDS copy (ASYNCcnt-tracked, bypasses VGPRs)
// LDS address comes from the vdst VGPR; low 32 bits of generic shared ptr = LDS off.
// ---------------------------------------------------------------------------
__device__ __forceinline__ void async_copy_b128(const void* g, const void* lds) {
  unsigned lds_off = (unsigned)(uintptr_t)lds;
  unsigned long long ga = (unsigned long long)(uintptr_t)g;
  asm volatile("global_load_async_to_lds_b128 %0, %1, off"
               :: "v"(lds_off), "v"(ga) : "memory");
}
__device__ __forceinline__ void wait_async_0() {
  asm volatile("s_wait_asynccnt 0x0" ::: "memory");
}
__device__ __forceinline__ void wait_async_6() {
  asm volatile("s_wait_asynccnt 0x6" ::: "memory");
}

// ---------------------------------------------------------------------------
// fp32 [K][N] -> bf16 [N][K]  (transposed weight conversion)
// ---------------------------------------------------------------------------
__global__ __launch_bounds__(256)
void convert_transpose_kernel(const float* __restrict__ in, bf16_t* __restrict__ out,
                              int K, int N) {
  size_t i = (size_t)blockIdx.x * 256 + threadIdx.x;
  if (i >= (size_t)N * K) return;
  int n = (int)(i / K);
  int k = (int)(i % K);
  out[i] = f2bf(in[(size_t)k * N + n]);
}

// ---------------------------------------------------------------------------
// LayerNorm over D=1024, one block per row, bf16 output
// ---------------------------------------------------------------------------
__global__ __launch_bounds__(256)
void layernorm_kernel(const float* __restrict__ x, const float* __restrict__ w,
                      const float* __restrict__ b, bf16_t* __restrict__ out) {
  __shared__ float red[256];
  const int row = blockIdx.x;
  const float* xr = x + (size_t)row * DD;
  const int tid = threadIdx.x;

  float s = 0.f;
  for (int i = tid; i < DD; i += 256) s += xr[i];
  red[tid] = s; __syncthreads();
  for (int off = 128; off > 0; off >>= 1) {
    if (tid < off) red[tid] += red[tid + off];
    __syncthreads();
  }
  float mean = red[0] / DD;
  __syncthreads();

  float v = 0.f;
  for (int i = tid; i < DD; i += 256) { float d = xr[i] - mean; v += d * d; }
  red[tid] = v; __syncthreads();
  for (int off = 128; off > 0; off >>= 1) {
    if (tid < off) red[tid] += red[tid + off];
    __syncthreads();
  }
  float rs = rsqrtf(red[0] / DD + 1e-5f);

  bf16_t* orow = out + (size_t)row * DD;
  for (int i = tid; i < DD; i += 256)
    orow[i] = f2bf((xr[i] - mean) * rs * w[i] + b[i]);
}

// ---------------------------------------------------------------------------
// Double-buffered bf16 WMMA GEMM: C[M][N] = A[M][K] * Bt[N][K]^T, fp32 acc.
// Block 256 thr = 8 waves (2x4); block tile 128x256, BK=32; wave tile 64x64
// (16 WMMA per stage). Tiles staged with GLOBAL_LOAD_ASYNC_TO_LDS_B128 and
// pipelined on ASYNCcnt (async loads complete in order).
// ---------------------------------------------------------------------------
enum { EPI_F32 = 0, EPI_RES_F32 = 1, EPI_GELU_BF16 = 2 };

#define GBM 128
#define GBN 256
#define GBK 32
#define GLDK 40   // padded LDS row stride in bf16 (80B: 16B-aligned, conflict-breaking)

template <int EPI>
__global__ __launch_bounds__(256)
void gemm_bf16_kernel(const bf16_t* __restrict__ A, const bf16_t* __restrict__ Bt,
                      const float* __restrict__ res,
                      float* __restrict__ Cf, bf16_t* __restrict__ Cb,
                      int M, int N, int K) {
  __shared__ bf16_t ldsA[2][GBM * GLDK];   // 2 x 10 KB
  __shared__ bf16_t ldsB[2][GBN * GLDK];   // 2 x 20 KB

  const int tid  = threadIdx.x;
  const int wave = tid >> 5;
  const int lane = tid & 31;
  const int g    = lane >> 4;
  const int lm   = lane & 15;
  const int wm   = wave >> 2;   // 0..1  (64 rows each)
  const int wn   = wave & 3;    // 0..3  (64 cols each)
  const int row0 = blockIdx.y * GBM;
  const int col0 = blockIdx.x * GBN;

  v8f acc[4][4];
#pragma unroll
  for (int i = 0; i < 4; ++i)
#pragma unroll
    for (int j = 0; j < 4; ++j) acc[i][j] = zero8();

  const int c4 = tid & 3;     // 16B chunk within a 64B (32-elem) row
  const int lr = tid >> 2;    // 0..63

  // stage issue: 2 async B128 for A (128 rows) + 4 for B (256 rows) per thread
  auto issue_stage = [&](int buf, int k0) {
#pragma unroll
    for (int p = 0; p < 2; ++p) {
      const int rr = lr + p * 64;
      async_copy_b128(A + (size_t)(row0 + rr) * K + k0 + c4 * 8,
                      &ldsA[buf][rr * GLDK + c4 * 8]);
    }
#pragma unroll
    for (int p = 0; p < 4; ++p) {
      const int rr = lr + p * 64;
      async_copy_b128(Bt + (size_t)(col0 + rr) * K + k0 + c4 * 8,
                      &ldsB[buf][rr * GLDK + c4 * 8]);
    }
  };

  issue_stage(0, 0);
  int cur = 0;
  for (int k0 = 0; k0 < K; k0 += GBK) {
    const bool hasnext = (k0 + GBK < K);
    if (hasnext) issue_stage(cur ^ 1, k0 + GBK);
    // async loads retire in order: <=6 outstanding => current stage landed
    if (hasnext) wait_async_6(); else wait_async_0();
    __syncthreads();

    v16bf af[4];
#pragma unroll
    for (int i = 0; i < 4; ++i)
      af[i] = load_frag_a(&ldsA[cur][(wm * 64 + i * 16 + lm) * GLDK], g, 0);
#pragma unroll
    for (int j = 0; j < 4; ++j) {
      v16bf bq = load_frag_b(&ldsB[cur][(wn * 64 + j * 16 + lm) * GLDK], g, 0);
#pragma unroll
      for (int i = 0; i < 4; ++i)
        acc[i][j] = wmma_bf16(af[i], bq, acc[i][j]);
    }
    __syncthreads();   // all waves done with buf[cur] before it is re-filled
    cur ^= 1;
  }

  // Epilogue: C layout lane holds (M = g*8 + r, N = lm) per 16x16 tile
#pragma unroll
  for (int i = 0; i < 4; ++i) {
#pragma unroll
    for (int j = 0; j < 4; ++j) {
      const int colg = col0 + wn * 64 + j * 16 + lm;
#pragma unroll
      for (int r = 0; r < 8; ++r) {
        const int rowg = row0 + wm * 64 + i * 16 + g * 8 + r;
        const size_t idx = (size_t)rowg * N + colg;
        float v = acc[i][j][r];
        if (EPI == EPI_RES_F32) {
          Cf[idx] = v + res[idx];
        } else if (EPI == EPI_GELU_BF16) {
          Cb[idx] = f2bf(0.5f * v * (1.0f + erff(v * 0.70710678118f)));
        } else {
          Cf[idx] = v;
        }
      }
    }
  }
}

// ---------------------------------------------------------------------------
// RoPE + split QKV: qkv fp32 [B*S][3D] -> q,k (roped), v bf16 [B][H][S][64]
// ---------------------------------------------------------------------------
__global__ __launch_bounds__(64)
void rope_split_kernel(const float* __restrict__ qkv,
                       const float* __restrict__ sinb, const float* __restrict__ cosb,
                       bf16_t* __restrict__ q, bf16_t* __restrict__ k,
                       bf16_t* __restrict__ v) {
  const int idx = blockIdx.x;                 // b*H*S + h*S + s
  const int s = idx % SS;
  const int h = (idx / SS) % HH;
  const int b = idx / (SS * HH);
  const int hd = threadIdx.x;                 // 0..63

  const float sn = sinb[s * HDIM + hd];
  const float cs = cosb[s * HDIM + hd];
  const size_t rowbase = (size_t)(b * SS + s) * (3 * DD);
  const size_t obase   = ((size_t)(b * HH + h) * SS + s) * HDIM + hd;

#pragma unroll
  for (int ch = 0; ch < 2; ++ch) {
    const size_t cbase = rowbase + (size_t)ch * DD + h * HDIM;
    const float xv = qkv[cbase + hd];
    const float xr = (hd < 32) ? -qkv[cbase + hd + 32] : qkv[cbase + hd - 32];
    const bf16_t o = f2bf(xv * cs + xr * sn);
    if (ch == 0) q[obase] = o; else k[obase] = o;
  }
  v[obase] = f2bf(qkv[rowbase + 2 * DD + h * HDIM + hd]);
}

// ---------------------------------------------------------------------------
// Flash-style causal attention. Grid: (S/64, B*H). Block: 128 thr = 4 waves,
// one 16-row q tile per wave. K chunk async-staged to LDS; V staged transposed;
// online softmax; both matmuls on WMMA. Output bf16 in [b][s][D] layout.
// ---------------------------------------------------------------------------
#define ALD 72

__global__ __launch_bounds__(128)
void attention_kernel(const bf16_t* __restrict__ Q, const bf16_t* __restrict__ K,
                      const bf16_t* __restrict__ V, bf16_t* __restrict__ O) {
  __shared__ bf16_t ldsK [64 * ALD];
  __shared__ bf16_t ldsVt[64 * ALD];
  __shared__ bf16_t ldsP [4 * 16 * ALD];

  const int bh   = blockIdx.y;       // 0..31
  const int qblk = blockIdx.x;       // 0..31
  const int tid  = threadIdx.x;
  const int wave = tid >> 5;
  const int lane = tid & 31;
  const int g    = lane >> 4;
  const int lm   = lane & 15;
  const size_t base = (size_t)bh * SS * HDIM;
  const int q0 = qblk * 64 + wave * 16;

  // Q fragments (A-matrix), loaded once from global
  v16bf qf[2];
  const bf16_t* qrow = Q + base + (size_t)(q0 + lm) * HDIM;
  qf[0] = load_frag_a(qrow, g, 0);
  qf[1] = load_frag_a(qrow, g, 1);

  v8f oacc[4];
#pragma unroll
  for (int j = 0; j < 4; ++j) oacc[j] = zero8();
  float mi[8], li[8];
#pragma unroll
  for (int r = 0; r < 8; ++r) { mi[r] = -INFINITY; li[r] = 0.f; }

  const int nchunk = qblk + 1;       // causal bound (chunks of 64 kv)
  const int c8 = tid & 7;
  const int lr = tid >> 3;           // 0..15

  for (int c = 0; c < nchunk; ++c) {
    const int kv0 = c * 64;
    // K chunk: async global->LDS (row-major); V chunk: transpose via VALU
#pragma unroll
    for (int p = 0; p < 4; ++p) {
      const int rr = lr + p * 16;
      async_copy_b128(K + base + (size_t)(kv0 + rr) * HDIM + c8 * 8,
                      ldsK + rr * ALD + c8 * 8);
      const v8bf vv = *(const v8bf*)(V + base + (size_t)(kv0 + rr) * HDIM + c8 * 8);
#pragma unroll
      for (int e = 0; e < 8; ++e) ldsVt[(c8 * 8 + e) * ALD + rr] = vv[e];
    }
    wait_async_0();
    __syncthreads();

    // S = Q * K^T  (16 x 64), fp32
    v8f sacc[4];
#pragma unroll
    for (int j = 0; j < 4; ++j) sacc[j] = zero8();
#pragma unroll
    for (int t = 0; t < 2; ++t) {
#pragma unroll
      for (int j = 0; j < 4; ++j) {
        v16bf kf = load_frag_b(ldsK + (j * 16 + lm) * ALD, g, t);
        sacc[j] = wmma_bf16(qf[t], kf, sacc[j]);
      }
    }

    // scale + causal mask + row max
    float mnew[8];
#pragma unroll
    for (int r = 0; r < 8; ++r) {
      const int qi = q0 + g * 8 + r;
      float mx = -INFINITY;
#pragma unroll
      for (int j = 0; j < 4; ++j) {
        const int kv = kv0 + j * 16 + lm;
        float x = sacc[j][r] * 0.125f;          // 1/sqrt(64)
        x = (kv <= qi) ? x : -INFINITY;
        sacc[j][r] = x;
        mx = fmaxf(mx, x);
      }
      mnew[r] = mx;
    }
#pragma unroll
    for (int off = 1; off < 16; off <<= 1)
#pragma unroll
      for (int r = 0; r < 8; ++r)
        mnew[r] = fmaxf(mnew[r], __shfl_xor(mnew[r], off, 32));

    // online softmax update
    float rowsum[8];
#pragma unroll
    for (int r = 0; r < 8; ++r) {
      const float m2 = fmaxf(mi[r], mnew[r]);
      const float alpha = __expf(mi[r] - m2);
      float rsum = 0.f;
#pragma unroll
      for (int j = 0; j < 4; ++j) {
        const float p = __expf(sacc[j][r] - m2);
        sacc[j][r] = p;
        rsum += p;
      }
      rowsum[r] = rsum;
      mi[r] = m2;
      li[r] = li[r] * alpha;
#pragma unroll
      for (int j = 0; j < 4; ++j) oacc[j][r] *= alpha;
    }
#pragma unroll
    for (int off = 1; off < 16; off <<= 1)
#pragma unroll
      for (int r = 0; r < 8; ++r)
        rowsum[r] += __shfl_xor(rowsum[r], off, 32);
#pragma unroll
    for (int r = 0; r < 8; ++r) li[r] += rowsum[r];

    // P (bf16) -> wave-private LDS region (same-wave RAW: hw counters)
    bf16_t* pbase = ldsP + wave * 16 * ALD;
#pragma unroll
    for (int j = 0; j < 4; ++j)
#pragma unroll
      for (int r = 0; r < 8; ++r)
        pbase[(g * 8 + r) * ALD + j * 16 + lm] = f2bf(sacc[j][r]);

    // O += P * V   (A = P 16x64, B = Vt columns contiguous in kv)
#pragma unroll
    for (int t = 0; t < 2; ++t) {
      v16bf pf = load_frag_a(pbase + lm * ALD, g, t);
#pragma unroll
      for (int j = 0; j < 4; ++j) {
        v16bf vf = load_frag_b(ldsVt + (j * 16 + lm) * ALD, g, t);
        oacc[j] = wmma_bf16(pf, vf, oacc[j]);
      }
    }
    __syncthreads();   // protect ldsK/ldsVt before next chunk
  }

  // write O / l  as bf16 into [b][s][D] layout
  const int b = bh >> 4;
  const int h = bh & 15;
#pragma unroll
  for (int j = 0; j < 4; ++j) {
#pragma unroll
    for (int r = 0; r < 8; ++r) {
      const int qi = q0 + g * 8 + r;
      const float val = oacc[j][r] / li[r];
      O[((size_t)(b * SS + qi)) * DD + h * HDIM + j * 16 + lm] = f2bf(val);
    }
  }
}

// ---------------------------------------------------------------------------
// Host launcher
// ---------------------------------------------------------------------------
extern "C" void kernel_launch(void* const* d_in, const int* in_sizes, int n_in,
                              void* d_out, int out_size, void* d_ws, size_t ws_size,
                              hipStream_t stream) {
  const float* x      = (const float*)d_in[0];
  // d_in[1] = mask (unused; causal computed inline)
  const float* sinb   = (const float*)d_in[2];
  const float* cosb   = (const float*)d_in[3];
  const float* ln1_w  = (const float*)d_in[4];
  const float* ln1_b  = (const float*)d_in[5];
  const float* w_qkv  = (const float*)d_in[6];
  const float* w_proj = (const float*)d_in[7];
  const float* ln2_w  = (const float*)d_in[8];
  const float* ln2_b  = (const float*)d_in[9];
  const float* w_fc1  = (const float*)d_in[10];
  const float* w_fc2  = (const float*)d_in[11];
  float* out = (float*)d_out;

  char* p = (char*)d_ws;
  auto alloc = [&](size_t bytes) -> void* {
    void* r = (void*)p;
    p += (bytes + 255) & ~(size_t)255;
    return r;
  };

  bf16_t* hx1    = (bf16_t*)alloc((size_t)MTOT * DD * 2);        // LN1 out
  bf16_t* wqkvT  = (bf16_t*)alloc((size_t)3 * DD * DD * 2);      // [3D][D]
  bf16_t* wprojT = (bf16_t*)alloc((size_t)DD * DD * 2);          // [D][D]
  bf16_t* wfc1T  = (bf16_t*)alloc((size_t)FFN * DD * 2);         // [FFN][D]
  bf16_t* wfc2T  = (bf16_t*)alloc((size_t)DD * FFN * 2);         // [D][FFN]
  float*  qkv    = (float*) alloc((size_t)MTOT * 3 * DD * 4);    // fp32 QKV
  bf16_t* qb     = (bf16_t*)alloc((size_t)BB * HH * SS * HDIM * 2);
  bf16_t* kb     = (bf16_t*)alloc((size_t)BB * HH * SS * HDIM * 2);
  bf16_t* vb     = (bf16_t*)alloc((size_t)BB * HH * SS * HDIM * 2);
  bf16_t* attb   = (bf16_t*)alloc((size_t)MTOT * DD * 2);        // attn out bf16
  float*  x2     = (float*) alloc((size_t)MTOT * DD * 4);        // x + proj
  bf16_t* hx2    = (bf16_t*)alloc((size_t)MTOT * DD * 2);        // LN2 out
  bf16_t* fc1g   = (bf16_t*)alloc((size_t)MTOT * FFN * 2);       // gelu(fc1)
  (void)ws_size; (void)in_sizes; (void)n_in; (void)out_size;

  // 1) transposed bf16 weights
  {
    size_t n;
    n = (size_t)3 * DD * DD;
    convert_transpose_kernel<<<dim3((unsigned)((n + 255) / 256)), 256, 0, stream>>>(w_qkv,  wqkvT,  DD,  3 * DD);
    n = (size_t)DD * DD;
    convert_transpose_kernel<<<dim3((unsigned)((n + 255) / 256)), 256, 0, stream>>>(w_proj, wprojT, DD,  DD);
    n = (size_t)FFN * DD;
    convert_transpose_kernel<<<dim3((unsigned)((n + 255) / 256)), 256, 0, stream>>>(w_fc1,  wfc1T,  DD,  FFN);
    n = (size_t)DD * FFN;
    convert_transpose_kernel<<<dim3((unsigned)((n + 255) / 256)), 256, 0, stream>>>(w_fc2,  wfc2T,  FFN, DD);
  }

  // 2) LN1
  layernorm_kernel<<<MTOT, 256, 0, stream>>>(x, ln1_w, ln1_b, hx1);

  // 3) QKV GEMM: [4096 x 3072] = hx1 * w_qkv
  gemm_bf16_kernel<EPI_F32><<<dim3(3 * DD / GBN, MTOT / GBM), 256, 0, stream>>>(
      hx1, wqkvT, nullptr, qkv, nullptr, MTOT, 3 * DD, DD);

  // 4) RoPE + split into per-head bf16 Q/K/V
  rope_split_kernel<<<BB * HH * SS, 64, 0, stream>>>(qkv, sinb, cosb, qb, kb, vb);

  // 5) causal flash attention (WMMA), output bf16 [b][s][D]
  attention_kernel<<<dim3(SS / 64, BB * HH), 128, 0, stream>>>(qb, kb, vb, attb);

  // 6) proj GEMM + residual: x2 = x + attb * w_proj
  gemm_bf16_kernel<EPI_RES_F32><<<dim3(DD / GBN, MTOT / GBM), 256, 0, stream>>>(
      attb, wprojT, x, x2, nullptr, MTOT, DD, DD);

  // 7) LN2
  layernorm_kernel<<<MTOT, 256, 0, stream>>>(x2, ln2_w, ln2_b, hx2);

  // 8) FC1 GEMM + exact GELU -> bf16
  gemm_bf16_kernel<EPI_GELU_BF16><<<dim3(FFN / GBN, MTOT / GBM), 256, 0, stream>>>(
      hx2, wfc1T, nullptr, nullptr, fc1g, MTOT, FFN, DD);

  // 9) FC2 GEMM + residual -> d_out
  gemm_bf16_kernel<EPI_RES_F32><<<dim3(DD / GBN, MTOT / GBM), 256, 0, stream>>>(
      fc1g, wfc2T, x2, out, nullptr, MTOT, DD, FFN);
}
